// Correlation_Module_29944511988246
// MI455X (gfx1250) — matrix operand order repaired and checked by reading, not compiled, not run on hardware
//
#include <hip/hip_runtime.h>

typedef __bf16 bf16_t;
typedef __attribute__((ext_vector_type(2)))  __bf16 v2bf;
typedef __attribute__((ext_vector_type(8)))  __bf16 v8bf;
typedef __attribute__((ext_vector_type(16))) __bf16 v16bf;
typedef __attribute__((ext_vector_type(8)))  float  v8f;

#define L_TOT   512
#define O_TOT   512
#define N_TOT   64
#define D_TOT   512
#define L_TILE  128
#define O_CHUNK 64
#define NTHREADS 512

// LDS row strides (elements), all 16B-aligned (stride*2 % 16 == 0)
#define XS 520   // Xs [L_TILE][XS]
#define US 520   // Us [O_CHUNK][US]   row-major U chunk
#define UT 72    // Ut [D_TOT][UT]     transposed U chunk
#define WS 72    // Ws [L_TILE][WS]    sigmoid weights

union V16 { v16bf v; v8bf h[2]; };

// ---- f32 -> bf16 conversion: prefer native packed convert ----
__device__ __forceinline__ v2bf cvt_pk(float a, float b) {
#if __has_builtin(__builtin_amdgcn_cvt_pk_bf16_f32)
    return __builtin_amdgcn_cvt_pk_bf16_f32(a, b);
#else
    v2bf r; r[0] = (__bf16)a; r[1] = (__bf16)b; return r;
#endif
}
__device__ __forceinline__ bf16_t cvt1(float a) {
#if __has_builtin(__builtin_amdgcn_cvt_pk_bf16_f32)
    v2bf t = __builtin_amdgcn_cvt_pk_bf16_f32(a, a);
    return t[0];
#else
    return (bf16_t)a;
#endif
}

// ---- tanh: prefer gfx1250 hardware V_TANH_F32 ----
__device__ __forceinline__ float fast_tanh(float v) {
#if __has_builtin(__builtin_amdgcn_tanhf)
    return __builtin_amdgcn_tanhf(v);
#elif __has_builtin(__builtin_amdgcn_tanh_f32)
    return __builtin_amdgcn_tanh_f32(v);
#else
    return tanhf(v);
#endif
}

__global__ __launch_bounds__(NTHREADS)
void corr_fused_kernel(const float* __restrict__ x,
                       const float* __restrict__ up,
                       float* __restrict__ out)
{
    extern __shared__ char smem_raw[];
    bf16_t* Xs = (bf16_t*)smem_raw;                 // 128*520
    bf16_t* Us = Xs + L_TILE * XS;                  // 64*520
    bf16_t* Ut = Us + O_CHUNK * US;                 // 512*72
    bf16_t* Ws = Ut + D_TOT * UT;                   // 128*72

    const int tid   = threadIdx.x;
    const int lane  = tid & 31;
    const int wave  = tid >> 5;
    const int n     = blockIdx.y;
    const int l0    = blockIdx.x * L_TILE;

    const int strip = wave >> 1;        // 0..7  (16-row strip)
    const int half  = wave & 1;         // 0..1
    const int m0    = strip * 16;
    const int lm    = lane & 15;
    const int lh    = lane >> 4;        // lane half-select

    const size_t row_stride = (size_t)N_TOT * D_TOT;   // 32768 floats

    // ---------------- load X tile (f32 -> bf16) into LDS -----------------
    {
        const float* xbase = x + (size_t)l0 * row_stride + (size_t)n * D_TOT;
        #pragma unroll 4
        for (int p = 0; p < (L_TILE * (D_TOT / 4)) / NTHREADS; ++p) { // 32 passes
            int idx = p * NTHREADS + tid;
            int row = idx >> 7;          // 0..127
            int c4  = idx & 127;         // float4 column
            float4 v = *(const float4*)(xbase + (size_t)row * row_stride + c4 * 4);
            v2bf p0 = cvt_pk(v.x, v.y);
            v2bf p1 = cvt_pk(v.z, v.w);
            uint2 q;
            q.x = __builtin_bit_cast(unsigned, p0);
            q.y = __builtin_bit_cast(unsigned, p1);
            *(uint2*)&Xs[row * XS + c4 * 4] = q;   // single ds_store_b64
        }
    }

    v8f acc[16] = {};                    // F accumulators: 16 tiles x (16x16)
    const float scale = 0.044194173824159216f;   // 1/sqrt(512)

    for (int oc = 0; oc < O_TOT; oc += O_CHUNK) {
        __syncthreads();   // Xs ready (1st iter); Us/Ut/Ws no longer read (later iters)

        // ------- load U chunk: row-major (Us) + transposed (Ut), bf16 -------
        {
            const float* ubase = up + (size_t)oc * row_stride + (size_t)n * D_TOT;
            #pragma unroll 4
            for (int p = 0; p < (O_CHUNK * (D_TOT / 4)) / NTHREADS; ++p) { // 16 passes
                int idx = p * NTHREADS + tid;
                int row = idx >> 7;      // 0..63
                int c4  = idx & 127;
                float4 v = *(const float4*)(ubase + (size_t)row * row_stride + c4 * 4);
                v2bf p0 = cvt_pk(v.x, v.y);
                v2bf p1 = cvt_pk(v.z, v.w);
                uint2 q;
                q.x = __builtin_bit_cast(unsigned, p0);
                q.y = __builtin_bit_cast(unsigned, p1);
                *(uint2*)&Us[row * US + c4 * 4] = q;
                int c = c4 * 4;
                Ut[(c + 0) * UT + row] = p0[0];
                Ut[(c + 1) * UT + row] = p0[1];
                Ut[(c + 2) * UT + row] = p1[0];
                Ut[(c + 3) * UT + row] = p1[1];
            }
        }
        __syncthreads();

        // ------- GEMM1: S[m0..+15, ob..+31] = X_strip @ U_chunk^T ----------
        // software-pipelined over K=D (16 steps of 32)
        v8f s0 = {}, s1 = {};
        const int ob = half * 32;
        {
            const bf16_t* xr  = &Xs[(m0 + lm) * XS + lh * 16];
            const bf16_t* ur0 = &Us[(ob + lm) * US + lh * 16];
            const bf16_t* ur1 = &Us[(ob + 16 + lm) * US + lh * 16];
            V16 a[2], b0[2], b1[2];
            a[0].h[0]  = *(const v8bf*)xr;         a[0].h[1]  = *(const v8bf*)(xr + 8);
            b0[0].h[0] = *(const v8bf*)ur0;        b0[0].h[1] = *(const v8bf*)(ur0 + 8);
            b1[0].h[0] = *(const v8bf*)ur1;        b1[0].h[1] = *(const v8bf*)(ur1 + 8);
            #pragma unroll
            for (int i = 0; i < 16; ++i) {
                const int cur = i & 1, nxt = cur ^ 1;
                if (i + 1 < 16) {
                    const int dn = (i + 1) * 32;
                    a[nxt].h[0]  = *(const v8bf*)(xr + dn);
                    a[nxt].h[1]  = *(const v8bf*)(xr + dn + 8);
                    b0[nxt].h[0] = *(const v8bf*)(ur0 + dn);
                    b0[nxt].h[1] = *(const v8bf*)(ur0 + dn + 8);
                    b1[nxt].h[0] = *(const v8bf*)(ur1 + dn);
                    b1[nxt].h[1] = *(const v8bf*)(ur1 + dn + 8);
                }
                s0 = __builtin_amdgcn_wmma_f32_16x16x32_bf16(false, a[cur].v, false, b0[cur].v,
                                                             (short)0, s0, false, false);
                s1 = __builtin_amdgcn_wmma_f32_16x16x32_bf16(false, a[cur].v, false, b1[cur].v,
                                                             (short)0, s1, false, false);
            }
        }

        // weights = sigmoid(S*scale) - 0.5 = 0.5*tanh(0.5*scale*S); store bf16
        const float hs = 0.5f * scale;
        #pragma unroll
        for (int r = 0; r < 8; ++r) {
            int row = m0 + r + lh * 8;
            float w0 = 0.5f * fast_tanh(hs * s0[r]);
            float w1 = 0.5f * fast_tanh(hs * s1[r]);
            Ws[row * WS + ob + lm]      = cvt1(w0);
            Ws[row * WS + ob + 16 + lm] = cvt1(w1);
        }
        __syncthreads();

        // ------- GEMM2: F[m0-strip, db..+255] += W_strip @ U_chunk ---------
        // ping-pong prefetch of B fragments across the 16 d-tiles
        const int db = half * 256;
        #pragma unroll
        for (int ko = 0; ko < O_CHUNK; ko += 32) {
            V16 a;
            const bf16_t* wr = &Ws[(m0 + lm) * WS + ko + lh * 16];
            a.h[0] = *(const v8bf*)wr;
            a.h[1] = *(const v8bf*)(wr + 8);

            V16 b[2];
            const bf16_t* u0 = &Ut[(db + lm) * UT + ko + lh * 16];
            b[0].h[0] = *(const v8bf*)u0;
            b[0].h[1] = *(const v8bf*)(u0 + 8);
            #pragma unroll
            for (int t = 0; t < 16; ++t) {
                if (t + 1 < 16) {
                    const bf16_t* un = &Ut[(db + (t + 1) * 16 + lm) * UT + ko + lh * 16];
                    b[(t + 1) & 1].h[0] = *(const v8bf*)un;
                    b[(t + 1) & 1].h[1] = *(const v8bf*)(un + 8);
                }
                acc[t] = __builtin_amdgcn_wmma_f32_16x16x32_bf16(false, a.v, false, b[t & 1].v,
                                                                 (short)0, acc[t], false, false);
            }
        }
    }

    // ---------------- store F tile (f32), coalesced per 16-lane group -------
    {
        float* obase = out + (size_t)l0 * row_stride + (size_t)n * D_TOT;
        const int db = half * 256;
        #pragma unroll
        for (int t = 0; t < 16; ++t) {
            int dcol = db + t * 16 + lm;
            #pragma unroll
            for (int r = 0; r < 8; ++r) {
                int row = m0 + r + lh * 8;
                obase[(size_t)row * row_stride + dcol] = acc[t][r];
            }
        }
    }
}

extern "C" void kernel_launch(void* const* d_in, const int* in_sizes, int n_in,
                              void* d_out, int out_size, void* d_ws, size_t ws_size,
                              hipStream_t stream) {
    (void)in_sizes; (void)n_in; (void)out_size; (void)d_ws; (void)ws_size;
    const float* x  = (const float*)d_in[0];
    const float* up = (const float*)d_in[1];
    float* out = (float*)d_out;

    size_t lds_bytes = (size_t)(L_TILE * XS + O_CHUNK * US + D_TOT * UT + L_TILE * WS)
                       * sizeof(bf16_t);   // 291,840 B <= 320 KB/WGP

    hipFuncSetAttribute((const void*)corr_fused_kernel,
                        hipFuncAttributeMaxDynamicSharedMemorySize, (int)lds_bytes);

    dim3 grid(L_TOT / L_TILE, N_TOT);   // (4, 64)
    dim3 block(NTHREADS);
    hipLaunchKernelGGL(corr_fused_kernel, grid, block, lds_bytes, stream, x, up, out);
}
